// CrossTransformerLayer_34772055228574
// MI455X (gfx1250) — matrix operand order repaired and checked by reading, not compiled, not run on hardware
//
#include <hip/hip_runtime.h>
#include <hip/hip_bf16.h>
#include <math.h>
#include <stdint.h>

typedef __attribute__((ext_vector_type(16))) _Float16 v16h;
typedef __attribute__((ext_vector_type(8)))  float    v8f;
typedef __attribute__((ext_vector_type(4)))  int      v4i;

#define DEV static __device__ __forceinline__

// --------------------------------------------------------------------------
// Optional gfx1250 async global->LDS staging (ASYNCcnt path). Fall back to
// register staging if the toolchain does not declare the builtins.
// Probe-derived prototype: (v4i addrspace(1)*, v4i addrspace(3)*, imm, imm).
// --------------------------------------------------------------------------
#if defined(__has_builtin)
#if __has_builtin(__builtin_amdgcn_global_load_async_to_lds_b128) && \
    __has_builtin(__builtin_amdgcn_s_wait_asynccnt)
#define USE_ASYNC_LDS 1
#endif
#endif
#ifndef USE_ASYNC_LDS
#define USE_ASYNC_LDS 0
#endif

#if USE_ASYNC_LDS
typedef __attribute__((address_space(1))) v4i gv4i;
typedef __attribute__((address_space(3))) v4i lv4i;
DEV void g2lds_b128(const void* g, void* l) {
  __builtin_amdgcn_global_load_async_to_lds_b128(
      (gv4i*)(uintptr_t)g, (lv4i*)(uint32_t)(uintptr_t)l, 0, 0);
}
DEV void wait_async_lds() { __builtin_amdgcn_s_wait_asynccnt(0); }
#endif

DEV v8f zero8() { v8f z = {0.f,0.f,0.f,0.f,0.f,0.f,0.f,0.f}; return z; }

// --------------------------------------------------------------------------
// WMMA fragment gathers (contiguous k = 16*half + s order, wave32). K-pairs are
// contiguous -> each fragment lowers to 2x ds_load_b128.
// --------------------------------------------------------------------------
DEV v16h load_a_frag(const _Float16* base, int m0, int k0, int ld, int lane) {
  const int half = (lane >> 4) & 1, mr = lane & 15;
  const unsigned* p = (const unsigned*)base;
  union { v16h h; unsigned u[8]; } f;
  const int row = (m0 + mr) * ld;
  const int kb  = k0 + half * 16;
#pragma unroll
  for (int j = 0; j < 4; ++j) f.u[j]     = p[(row + kb + 2*j) >> 1];
#pragma unroll
  for (int j = 0; j < 4; ++j) f.u[4 + j] = p[(row + 8 + kb + 2*j) >> 1];
  return f.h;
}

DEV v16h load_bt_frag(const _Float16* base, int n0, int k0, int ld, int lane) {
  const int half = (lane >> 4) & 1, nr = lane & 15;
  const unsigned* p = (const unsigned*)base;
  union { v16h h; unsigned u[8]; } f;
  const int row = (n0 + nr) * ld;
  const int kb  = k0 + half * 16;
#pragma unroll
  for (int j = 0; j < 8; ++j) f.u[j] = p[(row + kb + 2*j) >> 1];
  return f.h;
}

DEV v8f wmma_f16(v16h a, v16h b, v8f c) {
  return __builtin_amdgcn_wmma_f32_16x16x32_f16(false, a, false, b, (short)0, c,
                                                false, false);
}

// --------------------------------------------------------------------------
// f32 -> f16 elementwise (weights)
// --------------------------------------------------------------------------
__global__ void __launch_bounds__(256)
f32_to_f16_kernel(const float* __restrict__ a, _Float16* __restrict__ o, int n) {
  int i = blockIdx.x * 256 + threadIdx.x;
  if (i < n) o[i] = (_Float16)a[i];
}

// --------------------------------------------------------------------------
// LayerNorm (biased variance, eps=1e-5) -> f16, one block (8 waves) per row
// --------------------------------------------------------------------------
__global__ void __launch_bounds__(256)
ln_f16_kernel(const float* __restrict__ x, const float* __restrict__ g,
              const float* __restrict__ bta, _Float16* __restrict__ y, int Hn) {
  __shared__ float red[2][8];
  const int row = blockIdx.x, tid = threadIdx.x;
  const int wave = tid >> 5, lane = tid & 31;
  const float* xr = x + (size_t)row * Hn;
  float s = 0.f, s2 = 0.f;
  for (int c = tid; c < Hn; c += 256) { float v = xr[c]; s += v; s2 += v * v; }
#pragma unroll
  for (int o = 1; o < 32; o <<= 1) { s += __shfl_xor(s, o, 32); s2 += __shfl_xor(s2, o, 32); }
  if (lane == 0) { red[0][wave] = s; red[1][wave] = s2; }
  __syncthreads();
  float ts = 0.f, ts2 = 0.f;
#pragma unroll
  for (int i = 0; i < 8; ++i) { ts += red[0][i]; ts2 += red[1][i]; }
  const float mean = ts / (float)Hn;
  const float var  = ts2 / (float)Hn - mean * mean;
  const float inv  = rsqrtf(var + 1e-5f);
  _Float16* yr = y + (size_t)row * Hn;
  for (int c = tid; c < Hn; c += 256)
    yr[c] = (_Float16)((xr[c] - mean) * inv * g[c] + bta[c]);
}

// --------------------------------------------------------------------------
// WMMA GEMM: C[M,N] = A_f16[M,K] @ W_f16[K,N] + bias.
// Block tile 128x256 (8 waves, 2Mx4N), wave tile 64x64 = 16 WMMA per K-step.
// All 8 fragments of a K-step preloaded -> one dscnt wait, 16 WMMAs b2b.
// EPI: 0 = f32 out (+optional residual), 1 = f16 out scaled, 2 = GELU -> f16
// --------------------------------------------------------------------------
template <int EPI>
__global__ void __launch_bounds__(256, 1)
gemm_wmma_kernel(const _Float16* __restrict__ A, const _Float16* __restrict__ W,
                 const float* __restrict__ bias, const float* __restrict__ resid,
                 float* __restrict__ outf, _Float16* __restrict__ outh,
                 int M, int N, int K, float scale) {
  __shared__ __align__(16) _Float16 As[128 * 32];   // [m][k]
  __shared__ __align__(16) _Float16 Bs[256 * 32];   // [n][k] (K-major)
  const int tid = threadIdx.x;
  const int wave = tid >> 5, lane = tid & 31;
  const int bm = blockIdx.y * 128, bn = blockIdx.x * 256;
  const int wm = (wave & 1) * 64;        // 2 waves along M
  const int wn = (wave >> 1) * 64;       // 4 waves along N

  v8f acc[4][4];
#pragma unroll
  for (int mt = 0; mt < 4; ++mt)
#pragma unroll
    for (int nt = 0; nt < 4; ++nt) acc[mt][nt] = zero8();

  for (int k0 = 0; k0 < K; k0 += 32) {
    __syncthreads();
    // --- stage A tile 128x32 (direct copy; async path when available) ---
#pragma unroll
    for (int i = tid; i < 512; i += 256) {
      const int r = i >> 2, c = (i & 3) * 8;
#if USE_ASYNC_LDS
      g2lds_b128(&A[(size_t)(bm + r) * K + k0 + c], &As[r * 32 + c]);
#else
      *(float4*)&As[r * 32 + c] =
          *(const float4*)&A[(size_t)(bm + r) * K + k0 + c];
#endif
    }
    // --- stage W rows [k0,k0+32) x cols [bn,bn+256), transposed K-major ---
#pragma unroll
    for (int i = tid; i < 1024; i += 256) {
      const int r = i >> 5, c = (i & 31) * 8;
      float4 t = *(const float4*)&W[(size_t)(k0 + r) * N + bn + c];
      const _Float16* th = (const _Float16*)&t;
#pragma unroll
      for (int j = 0; j < 8; ++j) Bs[(c + j) * 32 + r] = th[j];
    }
    if (k0 + 32 < K) {  // cache hints for the next tiles
      __builtin_prefetch(&A[(size_t)(bm + (tid >> 1)) * K + k0 + 32], 0, 3);
      __builtin_prefetch(&W[(size_t)(k0 + 32 + (tid >> 4)) * N + bn + (tid & 15) * 16], 0, 3);
    }
#if USE_ASYNC_LDS
    wait_async_lds();
#endif
    __syncthreads();

    // --- preload all fragments, then 16 back-to-back WMMAs ---
    v16h bf[4], af[4];
#pragma unroll
    for (int nt = 0; nt < 4; ++nt)
      bf[nt] = load_bt_frag(Bs, wn + nt * 16, 0, 32, lane);
#pragma unroll
    for (int mt = 0; mt < 4; ++mt)
      af[mt] = load_a_frag(As, wm + mt * 16, 0, 32, lane);
#pragma unroll
    for (int mt = 0; mt < 4; ++mt)
#pragma unroll
      for (int nt = 0; nt < 4; ++nt)
        acc[mt][nt] = wmma_f16(af[mt], bf[nt], acc[mt][nt]);
  }

  // Epilogue: C VGPR r holds row (r + half*8), col = lane&15 within subtile
  const int half = lane >> 4, nr = lane & 15;
#pragma unroll
  for (int mt = 0; mt < 4; ++mt)
#pragma unroll
    for (int nt = 0; nt < 4; ++nt)
#pragma unroll
      for (int r = 0; r < 8; ++r) {
        const int row = bm + wm + mt * 16 + r + half * 8;
        const int col = bn + wn + nt * 16 + nr;
        const size_t idx = (size_t)row * N + col;
        float v = acc[mt][nt][r] + bias[col];
        if (EPI == 0) {
          if (resid) v += resid[idx];
          outf[idx] = v;
        } else if (EPI == 1) {
          outh[idx] = (_Float16)(v * scale);
        } else {  // exact GELU
          float gl = 0.5f * v * (1.0f + erff(v * 0.70710678118654752f));
          outh[idx] = (_Float16)gl;
        }
      }
}

// --------------------------------------------------------------------------
// Flash attention: block = (b, h, 64 q rows) = 4 waves x 16 rows.
// Per 64-key step: S = Q@K^T (WMMA) + bias, online softmax, O += P@V (WMMA).
// attn_bias (268 MB) is streamed exactly once; scores never hit HBM.
// --------------------------------------------------------------------------
__global__ void __launch_bounds__(128, 1)
attn_fa_kernel(const _Float16* __restrict__ q, const _Float16* __restrict__ k,
               const _Float16* __restrict__ v, const float* __restrict__ bias,
               _Float16* __restrict__ out) {
  const int S = 1024, H = 1024, NH = 16;
  __shared__ __align__(16) _Float16 kt[64 * 64];      // [key][d]
  __shared__ __align__(16) _Float16 vt[64 * 64];      // [d][key] (K-major for PV)
  __shared__ __align__(16) _Float16 pt[4][16 * 64];   // per-wave P tile [m][key]

  const int tid = threadIdx.x;
  const int wave = tid >> 5, lane = tid & 31;
  const int half = lane >> 4, lr = lane & 15;
  const int b = blockIdx.z, h = blockIdx.y;
  const int q0 = blockIdx.x * 64 + wave * 16;

  // Q fragments (held for the whole kernel; D^-0.5 already folded into q)
  const _Float16* qbase = q + ((size_t)(b * S + q0)) * H + h * 64;
  const v16h aq0 = load_a_frag(qbase, 0, 0, H, lane);
  const v16h aq1 = load_a_frag(qbase, 0, 32, H, lane);

  float mrow[8], lrow[8];
  v8f ao[4];
#pragma unroll
  for (int r = 0; r < 8; ++r) { mrow[r] = -3.0e38f; lrow[r] = 0.f; }
#pragma unroll
  for (int t = 0; t < 4; ++t) ao[t] = zero8();

  const float* bptr = bias + ((size_t)(b * NH + h)) * S * S;

  for (int kb = 0; kb < S; kb += 64) {
    __syncthreads();
    // Stage K [key][d] (async when available) and V transposed [d][key]
#pragma unroll
    for (int i = tid; i < 512; i += 128) {
      const int r = i >> 3, c = (i & 7) * 8;
      const size_t src = ((size_t)(b * S + kb + r)) * H + h * 64 + c;
#if USE_ASYNC_LDS
      g2lds_b128(&k[src], &kt[r * 64 + c]);
#else
      *(float4*)&kt[r * 64 + c] = *(const float4*)&k[src];
#endif
      float4 tv = *(const float4*)&v[src];
      const _Float16* th = (const _Float16*)&tv;
#pragma unroll
      for (int j = 0; j < 8; ++j) vt[(c + j) * 64 + r] = th[j];
    }
#if USE_ASYNC_LDS
    wait_async_lds();
#endif
    __syncthreads();

    // ---- scores: preload 8 K fragments, then 8 back-to-back WMMAs ----
    v16h bf[8];
#pragma unroll
    for (int t = 0; t < 4; ++t) {
      bf[2 * t]     = load_bt_frag(kt, t * 16, 0, 64, lane);
      bf[2 * t + 1] = load_bt_frag(kt, t * 16, 32, 64, lane);
    }
    v8f sc[4];
#pragma unroll
    for (int t = 0; t < 4; ++t) {
      sc[t] = wmma_f16(aq0, bf[2 * t], zero8());
      sc[t] = wmma_f16(aq1, bf[2 * t + 1], sc[t]);
    }
#pragma unroll
    for (int t = 0; t < 4; ++t)
#pragma unroll
      for (int r = 0; r < 8; ++r)
        sc[t][r] += bptr[(size_t)(q0 + r + half * 8) * S + kb + t * 16 + lr];

    // ---- online softmax (rows span the 16 lanes of this half) ----
#pragma unroll
    for (int r = 0; r < 8; ++r) {
      float mx = fmaxf(fmaxf(sc[0][r], sc[1][r]), fmaxf(sc[2][r], sc[3][r]));
#pragma unroll
      for (int o = 1; o < 16; o <<= 1) mx = fmaxf(mx, __shfl_xor(mx, o, 16));
      const float mn = fmaxf(mrow[r], mx);
      const float al = __expf(mrow[r] - mn);
      float rs = 0.f;
#pragma unroll
      for (int t = 0; t < 4; ++t) {
        float pv = __expf(sc[t][r] - mn);
        sc[t][r] = pv;
        rs += pv;
      }
#pragma unroll
      for (int o = 1; o < 16; o <<= 1) rs += __shfl_xor(rs, o, 16);
      lrow[r] = lrow[r] * al + rs;
      mrow[r] = mn;
#pragma unroll
      for (int t = 0; t < 4; ++t) ao[t][r] *= al;
    }

    // ---- P -> LDS (per-wave region; LDS ops in-order per wave) ----
#pragma unroll
    for (int t = 0; t < 4; ++t)
#pragma unroll
      for (int r = 0; r < 8; ++r)
        pt[wave][(r + half * 8) * 64 + t * 16 + lr] = (_Float16)sc[t][r];

    // ---- O += P @ V : preload 10 fragments, 8 back-to-back WMMAs ----
    const v16h af0 = load_a_frag(pt[wave], 0, 0, 64, lane);
    const v16h af1 = load_a_frag(pt[wave], 0, 32, 64, lane);
    v16h bv[8];
#pragma unroll
    for (int t = 0; t < 4; ++t) {
      bv[2 * t]     = load_bt_frag(vt, t * 16, 0, 64, lane);
      bv[2 * t + 1] = load_bt_frag(vt, t * 16, 32, 64, lane);
    }
#pragma unroll
    for (int t = 0; t < 4; ++t) {
      ao[t] = wmma_f16(af0, bv[2 * t], ao[t]);
      ao[t] = wmma_f16(af1, bv[2 * t + 1], ao[t]);
    }
  }

  // Final normalize (l==0 -> 0 reproduces the reference NaN cleanup), store
#pragma unroll
  for (int t = 0; t < 4; ++t)
#pragma unroll
    for (int r = 0; r < 8; ++r) {
      const float l = lrow[r];
      const float oval = (l > 0.f) ? ao[t][r] / l : 0.f;
      out[((size_t)(b * S + q0 + r + half * 8)) * H + h * 64 + t * 16 + lr] =
          (_Float16)oval;
    }
}

// --------------------------------------------------------------------------
// Host side
// --------------------------------------------------------------------------
extern "C" void kernel_launch(void* const* d_in, const int* in_sizes, int n_in,
                              void* d_out, int out_size, void* d_ws, size_t ws_size,
                              hipStream_t stream) {
  (void)in_sizes; (void)n_in; (void)out_size; (void)ws_size;
  const int Bn = 4, S = 1024, H = 1024, NH = 16, FF = 4096;
  const int M = Bn * S;  // 4096
  const size_t MH = (size_t)M * H, HH = (size_t)H * H;
  const size_t HF = (size_t)H * FF, MF = (size_t)M * FF;

  const float* x1   = (const float*)d_in[0];
  const float* x2   = (const float*)d_in[1];
  const float* abia = (const float*)d_in[2];
  const float* ln1g = (const float*)d_in[3];
  const float* ln1b = (const float*)d_in[4];
  const float* ln2g = (const float*)d_in[5];
  const float* ln2b = (const float*)d_in[6];
  const float* wq = (const float*)d_in[7];   const float* bq = (const float*)d_in[8];
  const float* wk = (const float*)d_in[9];   const float* bk = (const float*)d_in[10];
  const float* wv = (const float*)d_in[11];  const float* bv = (const float*)d_in[12];
  const float* wo = (const float*)d_in[13];  const float* bo = (const float*)d_in[14];
  const float* lnfg = (const float*)d_in[15];
  const float* lnfb = (const float*)d_in[16];
  const float* w1 = (const float*)d_in[17];  const float* b1 = (const float*)d_in[18];
  const float* w2 = (const float*)d_in[19];  const float* b2 = (const float*)d_in[20];

  char* wp = (char*)d_ws;
  _Float16* y1h  = (_Float16*)wp; wp += MH * 2;
  _Float16* y2h  = (_Float16*)wp; wp += MH * 2;
  _Float16* wq16 = (_Float16*)wp; wp += HH * 2;
  _Float16* wk16 = (_Float16*)wp; wp += HH * 2;
  _Float16* wv16 = (_Float16*)wp; wp += HH * 2;
  _Float16* wo16 = (_Float16*)wp; wp += HH * 2;
  _Float16* w116 = (_Float16*)wp; wp += HF * 2;
  _Float16* w216 = (_Float16*)wp; wp += HF * 2;
  _Float16* qh   = (_Float16*)wp; wp += MH * 2;
  _Float16* kh   = (_Float16*)wp; wp += MH * 2;
  _Float16* vh   = (_Float16*)wp; wp += MH * 2;
  _Float16* ah   = (_Float16*)wp; wp += MH * 2;
  float*    xres = (float*)wp;    wp += MH * 4;
  _Float16* yfh  = (_Float16*)wp; wp += MH * 2;
  _Float16* ffh  = (_Float16*)wp; wp += MF * 2;

  // 1) weights -> f16
  f32_to_f16_kernel<<<(int)((HH + 255) / 256), 256, 0, stream>>>(wq, wq16, (int)HH);
  f32_to_f16_kernel<<<(int)((HH + 255) / 256), 256, 0, stream>>>(wk, wk16, (int)HH);
  f32_to_f16_kernel<<<(int)((HH + 255) / 256), 256, 0, stream>>>(wv, wv16, (int)HH);
  f32_to_f16_kernel<<<(int)((HH + 255) / 256), 256, 0, stream>>>(wo, wo16, (int)HH);
  f32_to_f16_kernel<<<(int)((HF + 255) / 256), 256, 0, stream>>>(w1, w116, (int)HF);
  f32_to_f16_kernel<<<(int)((HF + 255) / 256), 256, 0, stream>>>(w2, w216, (int)HF);

  // 2) LayerNorms of x1 (k/v source) and x2 (q source)
  ln_f16_kernel<<<M, 256, 0, stream>>>(x1, ln1g, ln1b, y1h, H);
  ln_f16_kernel<<<M, 256, 0, stream>>>(x2, ln2g, ln2b, y2h, H);

  // 3) Q/K/V projections (D^-0.5 = 0.125 folded into Q)
  dim3 gHH(H / 256, M / 128);
  gemm_wmma_kernel<1><<<gHH, 256, 0, stream>>>(y2h, wq16, bq, nullptr, nullptr, qh,
                                               M, H, H, 0.125f);
  gemm_wmma_kernel<1><<<gHH, 256, 0, stream>>>(y1h, wk16, bk, nullptr, nullptr, kh,
                                               M, H, H, 1.0f);
  gemm_wmma_kernel<1><<<gHH, 256, 0, stream>>>(y1h, wv16, bv, nullptr, nullptr, vh,
                                               M, H, H, 1.0f);

  // 4) Attention (flash-style, bias streamed once)
  attn_fa_kernel<<<dim3(S / 64, NH, Bn), 128, 0, stream>>>(qh, kh, vh, abia, ah);

  // 5) Output projection + residual: xres = x2 + (attn @ wo + bo)
  gemm_wmma_kernel<0><<<gHH, 256, 0, stream>>>(ah, wo16, bo, x2, xres, nullptr,
                                               M, H, H, 1.0f);

  // 6) Final LN
  ln_f16_kernel<<<M, 256, 0, stream>>>(xres, lnfg, lnfb, yfh, H);

  // 7) FFN: gelu(yf @ w1 + b1) @ w2 + b2, out = xres + (...)
  gemm_wmma_kernel<2><<<dim3(FF / 256, M / 128), 256, 0, stream>>>(
      yfh, w116, b1, nullptr, nullptr, ffh, M, FF, H, 1.0f);
  gemm_wmma_kernel<0><<<gHH, 256, 0, stream>>>(ffh, w216, b2, xres, (float*)d_out,
                                               nullptr, M, H, FF, 1.0f);
}
